// CustomMamba_84370337562922
// MI455X (gfx1250) — matrix operand order repaired and checked by reading, not compiled, not run on hardware
//
#include <hip/hip_runtime.h>

// MI455X / gfx1250 fully-fused Mamba block.
// One workgroup (128 threads = 4 wave32) per sequence (B*N = 256 sequences).
// GEMMs via v_wmma_f32_16x16x32_f16 (f32 accumulate), scan in fp32 VALU.

typedef __attribute__((ext_vector_type(16))) _Float16 v16h;
typedef __attribute__((ext_vector_type(8)))  float    v8f;
typedef __attribute__((ext_vector_type(4)))  float    v4f;

union Frag16 { v16h h; v4f f[2]; };

__device__ __forceinline__ v8f wmma16(v16h a, v16h b, v8f c) {
  // 8 args: (neg_a, A, neg_b, B, c_mod, C, reuse_a, reuse_b)
  return __builtin_amdgcn_wmma_f32_16x16x32_f16(false, a, false, b, (short)0, c,
                                                false, false);
}

// A fragment: 16(M)x32(K) f16.  lane = g*16+m; halfs 0..7 <- K = kb+8g..+7,
// halfs 8..15 <- K = kb+16+8g..+7  (ISA 7.12.2, 16-bit A 16x32 layout).
__device__ __forceinline__ v16h load_frag_a(const _Float16* base, int row0,
                                            int ld, int kb) {
  const int lane = threadIdx.x & 31;
  const int m = lane & 15, g = lane >> 4;
  const _Float16* p = base + (row0 + m) * ld + kb + 8 * g;
  Frag16 fr;
  fr.f[0] = *(const v4f*)(p);
  fr.f[1] = *(const v4f*)(p + 16);
  return fr.h;
}

// B fragment: 32(K)x16(N) f16. lane = g*16+n; halfs 0..7 <- K = kb+8g..+7,
// halfs 8..15 <- K = kb+16+8g..+7 of column n (B^T stored row-major by output column).
__device__ __forceinline__ v16h load_frag_b(const _Float16* base, int col0,
                                            int ld, int kb) {
  const int lane = threadIdx.x & 31;
  const int n = lane & 15, g = lane >> 4;
  const _Float16* p = base + (col0 + n) * ld + kb + 8 * g;
  Frag16 fr;
  fr.f[0] = *(const v4f*)(p);
  fr.f[1] = *(const v4f*)(p + 16);
  return fr.h;
}

__device__ __forceinline__ float silu_f(float v) {
  return v / (1.f + __expf(-v));
}
__device__ __forceinline__ float softplus_f(float v) {
  return v > 20.f ? v : __logf(1.f + __expf(v));
}

// ---- LDS map (bytes), dynamic block, phase-lifetime aliased ----
// [0      ,16384 )  xh   f16[128][64]   phase1      | yh f16[128][128] phase4-5 (32768)
// [32768  ,65536 )  Wh   f16[256][64]   phase1
// [65536  ,131072)  upre f32[128][128]  phase1-2    | dt  f32[128][128] phase3.5-4
// [131072 ,163840)  zh   f16[128][128]  phase1-4
// [163840 ,229376)  u    f32[128][128]  phase2-4
// [229376 ,262144)  uh   f16[128][128]  phase2-3    | Woh f16[64][128]  phase3.5-5
// [262144 ,274432)  Wxh  f16[48][128]   phase0-3
// [274432 ,299008)  xdbl f32[128][48]   phase3-4
#define SMEM_BYTES 299008

__global__ __launch_bounds__(128, 1)
void mamba_fused_gfx1250(const float* __restrict__ x,
                         const float* __restrict__ W_in,
                         const float* __restrict__ conv_w,
                         const float* __restrict__ conv_b,
                         const float* __restrict__ W_x,
                         const float* __restrict__ W_dt,
                         const float* __restrict__ b_dt,
                         const float* __restrict__ A_log,
                         const float* __restrict__ Dp,
                         const float* __restrict__ W_out,
                         float* __restrict__ out) {
  extern __shared__ __align__(16) char smem[];
  _Float16* xh   = (_Float16*)(smem + 0);
  _Float16* yh   = (_Float16*)(smem + 0);
  _Float16* Wh   = (_Float16*)(smem + 32768);
  float*    upre = (float*)   (smem + 65536);
  float*    dtb  = (float*)   (smem + 65536);
  _Float16* zh   = (_Float16*)(smem + 131072);
  float*    us   = (float*)   (smem + 163840);
  _Float16* uh   = (_Float16*)(smem + 229376);
  _Float16* Woh  = (_Float16*)(smem + 229376);
  _Float16* Wxh  = (_Float16*)(smem + 262144);
  float*    xdbl = (float*)   (smem + 274432);

  const int tid  = threadIdx.x;
  const int wave = tid >> 5;
  const int lane = tid & 31;
  const int bn   = blockIdx.x;   // sequence index: seq[s] with s = b*64 + n
  const int b    = bn >> 6;
  const int n    = bn & 63;

  // x[b][t][n][f] at ((b*128+t)*64+n)*64+f
  const size_t xbase = ((size_t)b * 128u) * 4096u + (size_t)n * 64u;

  // ---- phase 0: stage x tile + weights into LDS (fp32 -> f16) ----
  __builtin_prefetch(x + xbase + (size_t)tid * 4096u, 0, 1);  // global_prefetch_b8
  for (int i = tid; i < 128 * 64; i += 128) {
    const int t = i >> 6, f = i & 63;
    xh[i] = (_Float16)x[xbase + (size_t)t * 4096u + (size_t)f];
  }
  for (int i = tid; i < 256 * 64; i += 128) Wh[i] = (_Float16)W_in[i];
  for (int i = tid; i < 48 * 128; i += 128) {
    const int j = i >> 7;
    Wxh[i] = (_Float16)((j < 36) ? W_x[i] : 0.f);  // pad rows 36..47 with zeros
  }
  __syncthreads();

  // ---- phase 1: xz = x @ W_in^T   (128 x 256, K=64) ----
  for (int tile = wave; tile < 128; tile += 4) {
    const int tm = tile >> 4;   // t block  0..7
    const int jn = tile & 15;   // col block 0..15
    v8f acc = {};
#pragma unroll
    for (int kb = 0; kb < 64; kb += 32) {
      const v16h a  = load_frag_a(xh, tm * 16, 64, kb);
      const v16h bf = load_frag_b(Wh, jn * 16, 64, kb);
      acc = wmma16(a, bf, acc);
    }
    const int g = lane >> 4, nn = lane & 15;
    if (jn < 8) {  // u (pre-conv) part, keep fp32
#pragma unroll
      for (int r = 0; r < 8; ++r)
        upre[(tm * 16 + 8 * g + r) * 128 + jn * 16 + nn] = acc[r];
    } else {       // z part, f16 is enough (only feeds silu gate)
#pragma unroll
      for (int r = 0; r < 8; ++r)
        zh[(tm * 16 + 8 * g + r) * 128 + (jn - 8) * 16 + nn] = (_Float16)acc[r];
    }
  }
  __syncthreads();

  // ---- phase 2: causal depthwise conv (taps=4) + bias + SiLU; thread=channel d ----
  {
    const int d = tid;
    const float cw0 = conv_w[d * 4 + 0], cw1 = conv_w[d * 4 + 1];
    const float cw2 = conv_w[d * 4 + 2], cw3 = conv_w[d * 4 + 3];
    const float cb = conv_b[d];
    float p1 = 0.f, p2 = 0.f, p3 = 0.f;  // u[t-1], u[t-2], u[t-3]
    for (int t = 0; t < 128; ++t) {
      const float cur = upre[t * 128 + d];
      const float v = cw0 * p3 + cw1 * p2 + cw2 * p1 + cw3 * cur + cb;
      p3 = p2; p2 = p1; p1 = cur;
      const float s = silu_f(v);
      us[t * 128 + d] = s;
      uh[t * 128 + d] = (_Float16)s;
    }
  }
  __syncthreads();

  // ---- phase 3: x_dbl = u @ W_x^T  (128 x 48[36 valid], K=128) ----
  for (int tile = wave; tile < 24; tile += 4) {
    const int tm = tile / 3;
    const int jn = tile % 3;
    v8f acc = {};
#pragma unroll
    for (int kb = 0; kb < 128; kb += 32) {
      const v16h a  = load_frag_a(uh, tm * 16, 128, kb);
      const v16h bf = load_frag_b(Wxh, jn * 16, 128, kb);
      acc = wmma16(a, bf, acc);
    }
    const int g = lane >> 4, nn = lane & 15;
#pragma unroll
    for (int r = 0; r < 8; ++r)
      xdbl[(tm * 16 + 8 * g + r) * 48 + jn * 16 + nn] = acc[r];
  }
  __syncthreads();

  // ---- phase 3.5: dt = softplus(x_dbl[:, :4] @ W_dt^T + b_dt) (K=4, VALU);
  //                 also stage W_out (overlays dead uh) ----
  {
    const int d = tid;
    const float w0 = W_dt[d * 4 + 0], w1 = W_dt[d * 4 + 1];
    const float w2 = W_dt[d * 4 + 2], w3 = W_dt[d * 4 + 3];
    const float bd = b_dt[d];
    for (int t = 0; t < 128; ++t) {
      const float* xr = xdbl + t * 48;
      const float s = bd + w0 * xr[0] + w1 * xr[1] + w2 * xr[2] + w3 * xr[3];
      dtb[t * 128 + d] = softplus_f(s);
    }
    for (int i = tid; i < 64 * 128; i += 128) Woh[i] = (_Float16)W_out[i];
  }
  __syncthreads();

  // ---- phase 4: selective scan, fp32, thread = channel d, h[16] in VGPRs ----
  {
    const int d = tid;
    float Arow[16];
#pragma unroll
    for (int s = 0; s < 16; ++s) Arow[s] = -__expf(A_log[d * 16 + s]);
    const float Dd = Dp[d];
    float h[16];
#pragma unroll
    for (int s = 0; s < 16; ++s) h[s] = 0.f;
    for (int t = 0; t < 128; ++t) {
      const float dtv = dtb[t * 128 + d];
      const float uv  = us[t * 128 + d];
      const float du  = dtv * uv;
      const float* xr = xdbl + t * 48;   // broadcast reads across lanes
      float y = 0.f;
#pragma unroll
      for (int s = 0; s < 16; ++s) {
        const float Bv = xr[4 + s];
        const float Cv = xr[20 + s];
        h[s] = __expf(dtv * Arow[s]) * h[s] + du * Bv;
        y += h[s] * Cv;
      }
      const float zv = (float)zh[t * 128 + d];
      const float yv = (y + uv * Dd) * silu_f(zv);
      yh[t * 128 + d] = (_Float16)yv;
    }
  }
  __syncthreads();

  // ---- phase 5: out = y @ W_out^T  (128 x 64, K=128), store to global ----
  for (int tile = wave; tile < 32; tile += 4) {
    const int tm = tile >> 2;
    const int fn = tile & 3;
    v8f acc = {};
#pragma unroll
    for (int kb = 0; kb < 128; kb += 32) {
      const v16h a  = load_frag_a(yh, tm * 16, 128, kb);
      const v16h bf = load_frag_b(Woh, fn * 16, 128, kb);
      acc = wmma16(a, bf, acc);
    }
    const int g = lane >> 4, nn = lane & 15;
#pragma unroll
    for (int r = 0; r < 8; ++r) {
      const int row = tm * 16 + 8 * g + r;  // t
      const int col = fn * 16 + nn;         // f
      out[(((size_t)b * 128 + row) * 64 + n) * 64 + col] = acc[r];
    }
  }
}

extern "C" void kernel_launch(void* const* d_in, const int* in_sizes, int n_in,
                              void* d_out, int out_size, void* d_ws,
                              size_t ws_size, hipStream_t stream) {
  (void)in_sizes; (void)n_in; (void)d_ws; (void)ws_size; (void)out_size;
  const float* x      = (const float*)d_in[0];
  const float* W_in   = (const float*)d_in[1];
  const float* conv_w = (const float*)d_in[2];
  const float* conv_b = (const float*)d_in[3];
  const float* W_x    = (const float*)d_in[4];
  const float* W_dt   = (const float*)d_in[5];
  const float* b_dt   = (const float*)d_in[6];
  const float* A_log  = (const float*)d_in[7];
  const float* Dp     = (const float*)d_in[8];
  const float* W_out  = (const float*)d_in[9];
  float* out = (float*)d_out;

  mamba_fused_gfx1250<<<dim3(256), dim3(128), SMEM_BYTES, stream>>>(
      x, W_in, conv_w, conv_b, W_x, W_dt, b_dt, A_log, Dp, W_out, out);
}
